// FtConditioner_26113401160070
// MI455X (gfx1250) — matrix operand order repaired and checked by reading, not compiled, not run on hardware
//
#include <hip/hip_runtime.h>
#include <hip/hip_bf16.h>
#include <math.h>

#define T_DIM 2048
#define B_DIM 8
#define S_DIM 512
#define D_DIM 512

typedef __attribute__((ext_vector_type(16))) __bf16         v16bf;
typedef __attribute__((ext_vector_type(8)))  float          v8f;
typedef __attribute__((ext_vector_type(8)))  unsigned short ushort8;
typedef __attribute__((ext_vector_type(4)))  unsigned int   uint4v;
typedef __attribute__((ext_vector_type(8)))  int            int8v;
typedef __attribute__((ext_vector_type(4)))  int            int4v;

union Frag16 { v16bf v; ushort8 h[2]; };

__device__ __forceinline__ unsigned short f32_to_bf16(float f) {
  unsigned u = __builtin_bit_cast(unsigned, f);
  unsigned r = u + 0x7FFFu + ((u >> 16) & 1u);   // round-to-nearest-even
  return (unsigned short)(r >> 16);
}

// ---------------------------------------------------------------------------
// Conversion kernels (one-time, bandwidth-trivial at 23.3 TB/s)
// ---------------------------------------------------------------------------
__global__ void conv_bf16_kernel(const float* __restrict__ src,
                                 unsigned short* __restrict__ dst, long n) {
  long i = (long)blockIdx.x * blockDim.x + threadIdx.x;
  long stride = (long)gridDim.x * blockDim.x;
  for (; i < n; i += stride) dst[i] = f32_to_bf16(src[i]);
}

// features [B,S,D] f32 -> fT [B,D,S] bf16  (B-fragments for the P*F GEMM)
__global__ void conv_featT_kernel(const float* __restrict__ f,
                                  unsigned short* __restrict__ ft) {
  const long n = (long)B_DIM * S_DIM * D_DIM;
  long i = (long)blockIdx.x * blockDim.x + threadIdx.x;
  long stride = (long)gridDim.x * blockDim.x;
  for (; i < n; i += stride) {
    int d = (int)(i % D_DIM);
    long bs = i / D_DIM;
    int s = (int)(bs % S_DIM);
    int b = (int)(bs / S_DIM);
    ft[((long)b * D_DIM + d) * S_DIM + s] = f32_to_bf16(f[i]);
  }
}

// W [D,2D] f32 -> wT [D,2D] bf16  (B-fragments for the output projection)
__global__ void conv_wT_kernel(const float* __restrict__ w,
                               unsigned short* __restrict__ wt) {
  const long n = (long)D_DIM * 2 * D_DIM;
  long i = (long)blockIdx.x * blockDim.x + threadIdx.x;
  long stride = (long)gridDim.x * blockDim.x;
  for (; i < n; i += stride) {
    int k = (int)(i % (2 * D_DIM));
    int r = (int)(i / (2 * D_DIM));
    wt[(long)r * (2 * D_DIM) + k] = f32_to_bf16(w[i]);
  }
}

// ---------------------------------------------------------------------------
// TDM staging of the 16x512 bf16 x-tile into LDS.
// D# per ISA ch.8: 2D tensor, data_size=3 (8B units), tile 128x16,
// row stride = B*D*2 bytes = 8192 B = 1024 units. type=2 ("image").
// ---------------------------------------------------------------------------
__device__ __forceinline__ void stage_xtile(unsigned short* xtile,
                                            const unsigned short* src_row0,
                                            int lane) {
#if __has_builtin(__builtin_amdgcn_tensor_load_to_lds)
  unsigned long long ga = (unsigned long long)(__UINTPTR_TYPE__)src_row0;
  unsigned lds_addr = (unsigned)(__UINTPTR_TYPE__)xtile;  // low 32b = LDS offset
  uint4v g0;
  g0[0] = 1u;                                   // count=1, user descriptor
  g0[1] = lds_addr;                             // lds_addr [63:32]
  g0[2] = (unsigned)(ga & 0xFFFFFFFFu);         // global_addr lo
  g0[3] = (unsigned)((ga >> 32) & 0x01FFFFFFu)  // global_addr hi (57b total)
          | (2u << 30);                         // type=2
  int8v g1;
  g1[0] = 0x30000;          // workgroup_mask=0, data_size=3 (8B)
  g1[1] = (int)(128u << 16);// tensor_dim0[15:0]=128 in bits[31:16]
  g1[2] = (int)(16u << 16); // tensor_dim0 hi=0 | tensor_dim1[15:0]=16
  g1[3] = (int)(128u << 16);// tensor_dim1 hi=0 | tile_dim0=128
  g1[4] = 16;               // tile_dim1=16, tile_dim2=0
  g1[5] = 1024;             // tensor_dim0_stride = 1024 (8B units)
  g1[6] = 0;
  g1[7] = 0;
  int4v z4 = {};
#if __clang_major__ >= 23
  int8v z8 = {};
  __builtin_amdgcn_tensor_load_to_lds(g0, g1, z4, z4, z8, 0);
#else
  __builtin_amdgcn_tensor_load_to_lds(g0, g1, z4, z4, 0);
#endif
  __builtin_amdgcn_s_wait_tensorcnt(0);
#else
  // Fallback: wave-strided b128 copy global(bf16) -> LDS
  for (int i = lane; i < 16 * 64; i += 32) {
    int m = i >> 6, c = i & 63;
    ((ushort8*)xtile)[i] =
        *(const ushort8*)(src_row0 + (long)m * B_DIM * D_DIM + c * 8);
  }
#endif
}

// ---------------------------------------------------------------------------
// Attention: one wave32 per (b, 16-row t-tile).
//   scores = Xtile[16xD] * F^T   (WMMA bf16, 16x32 output blocking)
//   softmax in f32 (wave shuffles), probs -> bf16 (reuse x-tile LDS)
//   x_res  = P[16xS] * F         (WMMA bf16, 16x32 output blocking)
// ---------------------------------------------------------------------------
__global__ __launch_bounds__(32) void attn_kernel(
    const unsigned short* __restrict__ xbf,   // [T*B,D] bf16 (row = t*B+b)
    const unsigned short* __restrict__ fbf,   // [B,S,D] bf16
    const unsigned short* __restrict__ fT,    // [B,D,S] bf16
    unsigned short* __restrict__ xres) {      // [T*B,D] bf16
  __shared__ unsigned short xtile[16 * 512];  // 16 KB: x tile, later probs
  __shared__ float sc[16 * 512];              // 32 KB: f32 scores / exp values

  const int lane = threadIdx.x;
  const int b = blockIdx.y;
  const int t0 = blockIdx.x * 16;

  stage_xtile(xtile, xbf + ((long)t0 * B_DIM + b) * D_DIM, lane);
  __syncthreads();

  const int M     = lane & 15;              // A-matrix row this lane owns
  const int koff  = (lane < 16) ? 0 : 8;    // A-matrix K sub-offset per spec
  const int nn    = lane & 15;              // C-matrix column
  const int mbase = (lane < 16) ? 0 : 8;    // C-matrix row base

  // ---- scores: 16x32 output tiles, K = D = 512 --------------------------
  for (int s0 = 0; s0 < S_DIM; s0 += 32) {
    v8f acc0 = {}, acc1 = {};
    for (int k0 = 0; k0 < D_DIM; k0 += 32) {
      Frag16 a, b0, b1;
      const unsigned short* ap = &xtile[M * 512 + k0 + koff];
      a.h[0] = *(const ushort8*)ap;
      a.h[1] = *(const ushort8*)(ap + 16);
      // B[k][n] = fbf[b][s0+n][k0+k]; lane<->N(s), elements<->K(d)
      const unsigned short* bp = &fbf[((long)b * S_DIM + s0 + nn) * D_DIM + k0 + koff];
      b0.h[0] = *(const ushort8*)bp;
      b0.h[1] = *(const ushort8*)(bp + 16);
      b1.h[0] = *(const ushort8*)(bp + 16 * D_DIM);
      b1.h[1] = *(const ushort8*)(bp + 16 * D_DIM + 16);
      acc0 = __builtin_amdgcn_wmma_f32_16x16x32_bf16(
          false, a.v, false, b0.v, (short)0, acc0, false, false);
      acc1 = __builtin_amdgcn_wmma_f32_16x16x32_bf16(
          false, a.v, false, b1.v, (short)0, acc1, false, false);
    }
    for (int r = 0; r < 8; ++r) {
      sc[(mbase + r) * 512 + s0 + nn] = acc0[r];
      sc[(mbase + r) * 512 + s0 + 16 + nn] = acc1[r];
    }
  }
  __syncthreads();

  // ---- softmax over S=512 per row, probs (bf16) overwrite xtile ---------
  for (int m = 0; m < 16; ++m) {
    float mx = -INFINITY;
    for (int s = lane; s < S_DIM; s += 32) mx = fmaxf(mx, sc[m * 512 + s]);
    for (int off = 16; off > 0; off >>= 1) mx = fmaxf(mx, __shfl_xor(mx, off, 32));
    float sum = 0.f;
    for (int s = lane; s < S_DIM; s += 32) {
      float e = __expf(sc[m * 512 + s] - mx);
      sc[m * 512 + s] = e;
      sum += e;
    }
    for (int off = 16; off > 0; off >>= 1) sum += __shfl_xor(sum, off, 32);
    float inv = 1.0f / sum;
    for (int s = lane; s < S_DIM; s += 32)
      xtile[m * 512 + s] = f32_to_bf16(sc[m * 512 + s] * inv);
  }
  __syncthreads();

  // ---- x_res = P @ F : 16x32 output tiles, K = S = 512 ------------------
  for (int d0 = 0; d0 < D_DIM; d0 += 32) {
    v8f acc0 = {}, acc1 = {};
    for (int k0 = 0; k0 < S_DIM; k0 += 32) {
      Frag16 a, b0, b1;
      const unsigned short* ap = &xtile[M * 512 + k0 + koff];
      a.h[0] = *(const ushort8*)ap;
      a.h[1] = *(const ushort8*)(ap + 16);
      // B[k][n] = fT[b][d0+n][k0+k]; lane<->N(d), elements<->K(s)
      const unsigned short* bp = &fT[((long)b * D_DIM + d0 + nn) * S_DIM + k0 + koff];
      b0.h[0] = *(const ushort8*)bp;
      b0.h[1] = *(const ushort8*)(bp + 16);
      b1.h[0] = *(const ushort8*)(bp + 16 * S_DIM);
      b1.h[1] = *(const ushort8*)(bp + 16 * S_DIM + 16);
      acc0 = __builtin_amdgcn_wmma_f32_16x16x32_bf16(
          false, a.v, false, b0.v, (short)0, acc0, false, false);
      acc1 = __builtin_amdgcn_wmma_f32_16x16x32_bf16(
          false, a.v, false, b1.v, (short)0, acc1, false, false);
    }
    for (int r = 0; r < 8; ++r) {
      int m = mbase + r;
      unsigned short* op = &xres[((long)(t0 + m) * B_DIM + b) * D_DIM + d0 + nn];
      op[0]  = f32_to_bf16(acc0[r]);
      op[16] = f32_to_bf16(acc1[r]);
    }
  }
}

// ---------------------------------------------------------------------------
// Output projection: out = x*W1^T + xres*W2^T + bias.
// M=16384, N=512, K=2x512. 8 waves/block; each wave owns a 32x64 register
// tile: 2 A-frags x 4 B-frags -> 8 WMMAs per K-step (load:wmma = 1.5:1).
// ---------------------------------------------------------------------------
__global__ __launch_bounds__(256) void linear_kernel(
    const unsigned short* __restrict__ xbf,   // [T*B,D] bf16
    const unsigned short* __restrict__ xres,  // [T*B,D] bf16
    const unsigned short* __restrict__ wT,    // [D,2D] bf16 (wT[n][k]=W[n][k])
    const float* __restrict__ bias,           // [D] f32
    float* __restrict__ out) {                // [T*B,D] f32
  const int lane = threadIdx.x & 31;
  const int wave = threadIdx.x >> 5;
  const int r0 = blockIdx.x * 32;             // 32-row tile
  const int n0 = wave * 64;                   // 64-col tile; 8 waves cover N

  const int M    = lane & 15;
  const int koff = (lane < 16) ? 0 : 8;

  v8f acc[2][4];
#pragma unroll
  for (int i = 0; i < 2; ++i)
#pragma unroll
    for (int j = 0; j < 4; ++j) acc[i][j] = (v8f){};

  const unsigned short* srcs[2] = {xbf, xres};
#pragma unroll
  for (int h = 0; h < 2; ++h) {
    const unsigned short* src = srcs[h];
    const int wOff = h * D_DIM;
    for (int k0 = 0; k0 < D_DIM; k0 += 32) {
      Frag16 a0, a1;
      const unsigned short* ap0 = &src[(long)(r0 + M) * D_DIM + k0 + koff];
      const unsigned short* ap1 = &src[(long)(r0 + 16 + M) * D_DIM + k0 + koff];
      a0.h[0] = *(const ushort8*)ap0;
      a0.h[1] = *(const ushort8*)(ap0 + 16);
      a1.h[0] = *(const ushort8*)ap1;
      a1.h[1] = *(const ushort8*)(ap1 + 16);
      const unsigned short* bp = &wT[(long)(n0 + (lane & 15)) * (2 * D_DIM) + wOff + k0 + koff];
      __builtin_prefetch(bp + 32 * D_DIM, 0, 0);  // weight rows n+16 (j=1 B-frag), same K step
#pragma unroll
      for (int j = 0; j < 4; ++j) {
        Frag16 bm;
        bm.h[0] = *(const ushort8*)(bp + 16 * j * 2 * D_DIM);
        bm.h[1] = *(const ushort8*)(bp + 16 * j * 2 * D_DIM + 16);
        acc[0][j] = __builtin_amdgcn_wmma_f32_16x16x32_bf16(
            false, a0.v, false, bm.v, (short)0, acc[0][j], false, false);
        acc[1][j] = __builtin_amdgcn_wmma_f32_16x16x32_bf16(
            false, a1.v, false, bm.v, (short)0, acc[1][j], false, false);
      }
    }
  }

  const int nn    = lane & 15;
  const int mbase = (lane < 16) ? 0 : 8;
#pragma unroll
  for (int j = 0; j < 4; ++j) {
    float bv = bias[n0 + 16 * j + nn];
#pragma unroll
    for (int i = 0; i < 2; ++i)
#pragma unroll
      for (int r = 0; r < 8; ++r)
        out[(long)(r0 + 16 * i + mbase + r) * D_DIM + n0 + 16 * j + nn] =
            acc[i][j][r] + bv;
  }
}

// ---------------------------------------------------------------------------
extern "C" void kernel_launch(void* const* d_in, const int* in_sizes, int n_in,
                              void* d_out, int out_size, void* d_ws, size_t ws_size,
                              hipStream_t stream) {
  const float* x        = (const float*)d_in[0];  // [T,B,D]
  const float* features = (const float*)d_in[1];  // [B,S,D]
  const float* W        = (const float*)d_in[2];  // [D,2D]
  const float* bias     = (const float*)d_in[3];  // [D]
  float* out = (float*)d_out;                     // [T,B,D]

  char* ws = (char*)d_ws;
  const size_t MB = 1024 * 1024;
  unsigned short* fbf  = (unsigned short*)(ws);             //  4 MB [B,S,D]
  unsigned short* fT   = (unsigned short*)(ws + 4 * MB);    //  4 MB [B,D,S]
  unsigned short* wT   = (unsigned short*)(ws + 8 * MB);    //  1 MB [D,2D]
  unsigned short* xbf  = (unsigned short*)(ws + 9 * MB);    // 16 MB [T*B,D]
  unsigned short* xres = (unsigned short*)(ws + 25 * MB);   // 16 MB [T*B,D]

  const long nFeat = (long)B_DIM * S_DIM * D_DIM;
  const long nX    = (long)T_DIM * B_DIM * D_DIM;

  conv_bf16_kernel<<<2048, 256, 0, stream>>>(features, fbf, nFeat);
  conv_bf16_kernel<<<4096, 256, 0, stream>>>(x, xbf, nX);
  conv_featT_kernel<<<2048, 256, 0, stream>>>(features, fT);
  conv_wT_kernel<<<1024, 256, 0, stream>>>(W, wT);

  dim3 ag(T_DIM / 16, B_DIM);                 // 128 x 8 t-tiles
  attn_kernel<<<ag, 32, 0, stream>>>(xbf, fbf, fT, xres);

  linear_kernel<<<(T_DIM * B_DIM) / 32, 256, 0, stream>>>(xbf, xres, wT, bias, out);
}
